// KernelUpdateHead_61667140436661
// MI455X (gfx1250) — compile-verified, matrix operand order checked by reading, not run on hardware
//
#include <hip/hip_runtime.h>
#include <hip/hip_bf16.h>

// ---------------- problem constants ----------------
constexpr int N_ = 2, P_ = 100, C_ = 256, H_ = 128, W_ = 192;
constexpr int HW_ = H_ * W_;            // 24576
constexpr int KK_ = 9;
constexpr int E_  = C_ * KK_;           // 2304
constexpr int NH_ = 8, DH_ = E_ / NH_;  // 288
constexpr int FFN_ = 2048;
constexpr int NP_ = N_ * P_;            // 200
constexpr int R_  = NP_ * KK_;          // 1800
constexpr long NM_ = (long)N_ * P_ * HW_; // new_masks elements

// ---------------- vector types ----------------
typedef _Float16 v16h __attribute__((ext_vector_type(16)));
typedef _Float16 v8h  __attribute__((ext_vector_type(8)));
typedef float    v8f  __attribute__((ext_vector_type(8)));

__device__ __forceinline__ v16h frag_ld(const _Float16* p) {
  v8h lo = *(const v8h*)(p);
  v8h hi = *(const v8h*)(p + 8);
  v16h r;
#pragma unroll
  for (int i = 0; i < 8; ++i) { r[i] = lo[i]; r[i + 8] = hi[i]; }
  return r;
}

// LDS offset of a __shared__ object: per ISA 10.2 the low 32 bits of the
// generic address of an LDS location are the LDS byte offset.
__device__ __forceinline__ unsigned lds_off(const void* p) {
  return (unsigned)(unsigned long long)p;
}

// Async global->LDS copies (CDNA5 ASYNCcnt path), per CDNA5 ISA.
__device__ __forceinline__ void async_ld_b32(unsigned ldsaddr, const float* gaddr) {
  asm volatile("global_load_async_to_lds_b32 %0, %1, off"
               :: "v"(ldsaddr), "v"(gaddr) : "memory");
}
__device__ __forceinline__ void async_ld_b128(unsigned ldsaddr, const float* gaddr) {
  asm volatile("global_load_async_to_lds_b128 %0, %1, off"
               :: "v"(ldsaddr), "v"(gaddr) : "memory");
}
__device__ __forceinline__ void wait_async0() {
  asm volatile("s_wait_asynccnt 0" ::: "memory");
}

// =====================================================================
// Generic NT WMMA GEMM:  C[M,Nc] = act(A[M,K] @ B[Nc,K]^T + bias)
// Requirements: K % 64 == 0, Nc % 32 == 0. M ragged (guarded).
// Block 256 threads = 8 waves, tile 64(M) x 32(N), K_BLK = 64.
// =====================================================================
__global__ __launch_bounds__(256) void gemm_nt_wmma(
    const float* __restrict__ A, long lda, long sAb,
    const float* __restrict__ B, long ldb, long sBb,
    const float* __restrict__ bias,
    float* __restrict__ C, long ldc, long sCb,
    int M, int Nc, int K, int act /*0 none, 1 relu*/) {
  constexpr int LD = 72;  // halves per LDS row (64 + 8 pad, keeps 16B align)
  __shared__ __align__(16) _Float16 As[64 * LD];
  __shared__ __align__(16) _Float16 Bs[32 * LD];

  const int nb = blockIdx.x * 32;
  const int mb = blockIdx.y * 64;
  A += (long)blockIdx.z * sAb;
  B += (long)blockIdx.z * sBb;
  C += (long)blockIdx.z * sCb;

  const int tid = threadIdx.x, lane = tid & 31, w = tid >> 5;
  const int mi = (w & 3) * 16;  // 4 M-subtiles
  const int ni = (w >> 2) * 16; // 2 N-subtiles
  v8f acc;
#pragma unroll
  for (int i = 0; i < 8; ++i) acc[i] = 0.f;

  const int arow = tid >> 2;       // 0..63
  const int acol = (tid & 3) * 16; // 0,16,32,48
  const int brow = tid >> 3;       // 0..31
  const int bcol = (tid & 7) * 8;  // 0..56

  const int kb = (lane >> 4) * 16;
  const int ar = mi + (lane & 15);
  const int bc = ni + (lane & 15);

  for (int k0 = 0; k0 < K; k0 += 64) {
    // stage A tile 64x64 (f32 -> f16), 16 elems/thread
    {
      const int gm = mb + arow;
      v8h h0, h1;
      if (gm < M) {
        const float* s = A + (long)gm * lda + k0 + acol;
        __builtin_prefetch(s + 64, 0, 0);  // global_prefetch_b8 next K chunk
#pragma unroll
        for (int i = 0; i < 8; ++i) h0[i] = (_Float16)s[i];
#pragma unroll
        for (int i = 0; i < 8; ++i) h1[i] = (_Float16)s[8 + i];
      } else {
#pragma unroll
        for (int i = 0; i < 8; ++i) { h0[i] = (_Float16)0.f; h1[i] = (_Float16)0.f; }
      }
      *(v8h*)&As[arow * LD + acol]     = h0;
      *(v8h*)&As[arow * LD + acol + 8] = h1;
    }
    // stage B tile 32x64, 8 elems/thread (Nc % 32 == 0 -> always valid)
    {
      const float* s = B + (long)(nb + brow) * ldb + k0 + bcol;
      v8h h0;
#pragma unroll
      for (int i = 0; i < 8; ++i) h0[i] = (_Float16)s[i];
      *(v8h*)&Bs[brow * LD + bcol] = h0;
    }
    __syncthreads();
#pragma unroll
    for (int ks = 0; ks < 64; ks += 32) {
      v16h af = frag_ld(&As[ar * LD + ks + kb]);
      v16h bf = frag_ld(&Bs[bc * LD + ks + kb]);
      acc = __builtin_amdgcn_wmma_f32_16x16x32_f16(false, af, false, bf,
                                                   (short)0, acc, false, false);
    }
    __syncthreads();
  }

  const int cn = nb + ni + (lane & 15);
  const int m0 = mb + mi + (lane >> 4) * 8;
  const float bv = bias ? bias[cn] : 0.f;
#pragma unroll
  for (int v = 0; v < 8; ++v) {
    const int m = m0 + v;
    if (m < M) {
      float val = acc[v] + bv;
      if (act == 1) val = fmaxf(val, 0.f);
      C[(long)m * ldc + cn] = val;
    }
  }
}

// =====================================================================
// Implicit-GEMM 3x3 SAME conv via WMMA.
// out[n][p][y][x] = sum_{c,dy,dx} x[n][c][y+dy-1][x+dx-1] * ker[n][p][c][dy*3+dx]
// Block: 32(P) x 64(pixels in one image row). K-loop: 9 taps x C (chunks of 64).
// =====================================================================
__global__ __launch_bounds__(256) void conv3x3_wmma(
    const float* __restrict__ x, const float* __restrict__ ker,
    float* __restrict__ out) {
  constexpr int LD = 72;
  __shared__ __align__(16) _Float16 As[32 * LD]; // kernel tile: 32p x 64c
  __shared__ __align__(16) _Float16 Bs[64 * LD]; // image tile: 64px x 64c

  const int n = blockIdx.z;
  const int p0 = blockIdx.y * 32;
  const int yrow = blockIdx.x / 3;
  const int x0 = (blockIdx.x % 3) * 64;

  const int tid = threadIdx.x, lane = tid & 31, w = tid >> 5;
  const int mi = (w & 1) * 16;  // 2 P-subtiles
  const int ni = (w >> 1) * 16; // 4 px-subtiles
  v8f acc;
#pragma unroll
  for (int i = 0; i < 8; ++i) acc[i] = 0.f;

  const int kb = (lane >> 4) * 16;
  const int ar = mi + (lane & 15);
  const int bc = ni + (lane & 15);

  for (int tap = 0; tap < 9; ++tap) {
    const int dy = tap / 3 - 1, dx = tap % 3 - 1;
    const int yin = yrow + dy;
    if (yin < 0 || yin >= H_) continue;  // zero padding contributes nothing
    for (int kc = 0; kc < C_; kc += 64) {
      // A: ker[n][p0+pl][kc+cs+i][tap], stride 9 across c
      {
        const int pl = tid >> 3;        // 0..31
        const int cs = (tid & 7) * 8;   // 0..56
        const int gp = p0 + pl;
        v8h hv;
        if (gp < P_) {
          const float* s = ker + ((long)(n * P_ + gp) * C_ + kc + cs) * KK_ + tap;
#pragma unroll
          for (int i = 0; i < 8; ++i) hv[i] = (_Float16)s[(long)i * KK_];
        } else {
#pragma unroll
          for (int i = 0; i < 8; ++i) hv[i] = (_Float16)0.f;
        }
        *(v8h*)&As[pl * LD + cs] = hv;
      }
      // B: x[n][kc+cl][yin][x0+dx+js+i] stored transposed Bs[px][c]
      {
        const int cl = tid >> 2;        // 0..63
        const int js = (tid & 3) * 16;  // 0..48
        const float* s = x + ((long)(n * C_ + kc + cl) * H_ + yin) * W_;
#pragma unroll
        for (int i = 0; i < 16; ++i) {
          const int xx = x0 + dx + js + i;
          const float v = (xx >= 0 && xx < W_) ? s[xx] : 0.f;
          Bs[(js + i) * LD + cl] = (_Float16)v;
        }
      }
      __syncthreads();
#pragma unroll
      for (int ks = 0; ks < 64; ks += 32) {
        v16h af = frag_ld(&As[ar * LD + ks + kb]);
        v16h bf = frag_ld(&Bs[bc * LD + ks + kb]);
        acc = __builtin_amdgcn_wmma_f32_16x16x32_f16(false, af, false, bf,
                                                     (short)0, acc, false, false);
      }
      __syncthreads();
    }
  }

  const int px = x0 + ni + (lane & 15);
  const int pr0 = p0 + mi + (lane >> 4) * 8;
#pragma unroll
  for (int v = 0; v < 8; ++v) {
    const int p = pr0 + v;
    if (p < P_)
      out[((long)(n * P_ + p) * H_ + yrow) * W_ + px] = acc[v];
  }
}

// =====================================================================
// Softmax over the P axis of mask_preds (N,P,H,W), per pixel.
// =====================================================================
__global__ __launch_bounds__(256) void softmax_p_kernel(
    const float* __restrict__ mp, float* __restrict__ sig) {
  const long idx = (long)blockIdx.x * blockDim.x + threadIdx.x;
  if (idx >= (long)N_ * HW_) return;
  const int n = (int)(idx / HW_);
  const int pix = (int)(idx % HW_);
  const float* src = mp + (long)n * P_ * HW_ + pix;
  float mx = -1e30f;
  for (int p = 0; p < P_; ++p) mx = fmaxf(mx, src[(long)p * HW_]);
  float s = 0.f;
  for (int p = 0; p < P_; ++p) s += __expf(src[(long)p * HW_] - mx);
  const float inv = 1.f / s;
  float* dst = sig + (long)n * P_ * HW_ + pix;
  for (int p = 0; p < P_; ++p)
    dst[(long)p * HW_] = __expf(src[(long)p * HW_] - mx) * inv;
}

// =====================================================================
// LayerNorm over last dim. out = mode(LN(x + res) * g + b)
// mode: 0 none, 1 sigmoid, 2 relu. One block per row.
// The row is staged global->LDS once with async-LDS DMA (ASYNCcnt) and
// consumed twice from LDS (mean/var pass + normalize pass).
// =====================================================================
__global__ __launch_bounds__(256) void layernorm_kernel(
    const float* __restrict__ xin, long ldx,
    const float* __restrict__ res, long ldr,
    const float* __restrict__ g, const float* __restrict__ b,
    float* __restrict__ out, long ldo, int Cdim, int mode) {
  __shared__ __align__(16) float rowbuf[E_];  // Cdim <= 2304
  __shared__ float sbuf[256], qbuf[256];
  const int row = blockIdx.x;
  const float* xr = xin + (long)row * ldx;
  const float* rr = res ? res + (long)row * ldr : nullptr;

  // async DMA the row into LDS (no VGPR round trip)
  for (int c = threadIdx.x; c < Cdim; c += 256)
    async_ld_b32(lds_off(&rowbuf[c]), xr + c);
  wait_async0();
  __syncthreads();

  float s = 0.f, q = 0.f;
  for (int c = threadIdx.x; c < Cdim; c += 256) {
    const float v = rowbuf[c] + (rr ? rr[c] : 0.f);
    s += v; q += v * v;
  }
  sbuf[threadIdx.x] = s; qbuf[threadIdx.x] = q;
  __syncthreads();
  for (int off = 128; off > 0; off >>= 1) {
    if (threadIdx.x < off) {
      sbuf[threadIdx.x] += sbuf[threadIdx.x + off];
      qbuf[threadIdx.x] += qbuf[threadIdx.x + off];
    }
    __syncthreads();
  }
  const float mean = sbuf[0] / Cdim;
  const float var  = qbuf[0] / Cdim - mean * mean;
  const float inv  = rsqrtf(var + 1e-5f);
  for (int c = threadIdx.x; c < Cdim; c += 256) {
    const float v = rowbuf[c] + (rr ? rr[c] : 0.f);
    float y = (v - mean) * inv * g[c] + b[c];
    if (mode == 1) y = 1.f / (1.f + __expf(-y));
    else if (mode == 2) y = fmaxf(y, 0.f);
    out[(long)row * ldo + c] = y;
  }
}

// ---------------- elementwise helpers ----------------
// gate[r][c] = inp[r][c] * params[np][c]   (r = np*9+kk; inp/params ld=512)
__global__ __launch_bounds__(256) void gate_kernel(
    const float* __restrict__ inp, const float* __restrict__ params,
    float* __restrict__ gate) {
  const long i = (long)blockIdx.x * 256 + threadIdx.x;
  if (i >= (long)R_ * C_) return;
  const int c = (int)(i % C_);
  const long r = i / C_;
  const long np = r / KK_;
  gate[i] = inp[r * (2 * C_) + c] * params[np * (2 * C_) + c];
}

// feat = ug * pout(broadcast over kk) + ig * iout
__global__ __launch_bounds__(256) void feat_combine_kernel(
    const float* __restrict__ ug, const float* __restrict__ ig,
    const float* __restrict__ pout, const float* __restrict__ iout,
    float* __restrict__ feat) {
  const long i = (long)blockIdx.x * 256 + threadIdx.x;
  if (i >= (long)R_ * C_) return;
  const int c = (int)(i % C_);
  const long np = (i / C_) / KK_;
  feat[i] = ug[i] * pout[np * C_ + c] + ig[i] * iout[i];
}

// obj[(p*N+n)*E + kk*C + c] = feat[((n*P+p)*KK+kk)*C + c]
__global__ __launch_bounds__(256) void pack_obj_kernel(
    const float* __restrict__ feat, float* __restrict__ obj) {
  const long i = (long)blockIdx.x * 256 + threadIdx.x;
  if (i >= (long)R_ * C_) return;
  const int c = (int)(i % C_);
  long t = i / C_;
  const int kk = (int)(t % KK_);
  const long np = t / KK_;
  const int n = (int)(np / P_), p = (int)(np % P_);
  obj[(long)(p * N_ + n) * E_ + kk * C_ + c] = feat[i];
}

// objr[((n*P+p)*KK+kk)*C + c] = obj[(p*N+n)*E + kk*C + c]
__global__ __launch_bounds__(256) void unpack_obj_kernel(
    const float* __restrict__ obj, float* __restrict__ objr) {
  const long i = (long)blockIdx.x * 256 + threadIdx.x;
  if (i >= (long)R_ * C_) return;
  const int c = (int)(i % C_);
  long t = i / C_;
  const int kk = (int)(t % KK_);
  const long np = t / KK_;
  const int n = (int)(np / P_), p = (int)(np % P_);
  objr[i] = obj[(long)(p * N_ + n) * E_ + kk * C_ + c];
}

// (np,kk,c) -> (np,c,kk): dst[(np*C+c)*KK+kk] = src[(np*KK+kk)*C+c]
__global__ __launch_bounds__(256) void kkc_to_ckk_kernel(
    const float* __restrict__ src, float* __restrict__ dst) {
  const long i = (long)blockIdx.x * 256 + threadIdx.x;
  if (i >= (long)R_ * C_) return;
  const int c = (int)(i % C_);
  long t = i / C_;
  const int kk = (int)(t % KK_);
  const long np = t / KK_;
  dst[(np * C_ + c) * KK_ + kk] = src[i];
}

// (np,c,kk) -> (np,kk,c): dst[(np*KK+kk)*C+c] = src[(np*C+c)*KK+kk]
__global__ __launch_bounds__(256) void ckk_to_kkc_kernel(
    const float* __restrict__ src, float* __restrict__ dst) {
  const long i = (long)blockIdx.x * 256 + threadIdx.x;
  if (i >= (long)R_ * C_) return;
  const int c = (int)(i % C_);
  long t = i / C_;
  const int kk = (int)(t % KK_);
  const long np = t / KK_;
  dst[i] = src[(np * C_ + c) * KK_ + kk];
}

// =====================================================================
// Multi-head self attention over P (tiny: P=100, NH=8, N=2).
// qkv rows ordered (p*N+n), columns [q|k|v] each E wide.
// One block per (q,h,n). q-vector staged to LDS with async b128 DMA;
// 1/sqrt(DH) applied on the scores (equivalent).
// =====================================================================
__global__ __launch_bounds__(128) void attn_kernel(
    const float* __restrict__ qkv, float* __restrict__ o) {
  const int q = blockIdx.x, h = blockIdx.y, n = blockIdx.z;
  __shared__ __align__(16) float qs[DH_];
  __shared__ float sc[P_];
  __shared__ float wgt[P_];
  const float scale = rsqrtf((float)DH_);
  const float* qrow = qkv + (long)(q * N_ + n) * (3 * E_) + h * DH_;
  // 288 floats = 72 lanes x 16B async DMA into LDS
  if (threadIdx.x < DH_ / 4)
    async_ld_b128(lds_off(&qs[threadIdx.x * 4]), qrow + threadIdx.x * 4);
  wait_async0();
  __syncthreads();
  for (int k = threadIdx.x; k < P_; k += 128) {
    const float* krow = qkv + (long)(k * N_ + n) * (3 * E_) + E_ + h * DH_;
    float a = 0.f;
    for (int d = 0; d < DH_; ++d) a += qs[d] * krow[d];
    sc[k] = a * scale;
  }
  __syncthreads();
  float mx = -1e30f;
  for (int k = 0; k < P_; ++k) mx = fmaxf(mx, sc[k]);
  float sum = 0.f;
  for (int k = 0; k < P_; ++k) sum += __expf(sc[k] - mx);
  const float inv = 1.f / sum;
  for (int k = threadIdx.x; k < P_; k += 128) wgt[k] = __expf(sc[k] - mx) * inv;
  __syncthreads();
  for (int d = threadIdx.x; d < DH_; d += 128) {
    float a = 0.f;
    for (int k = 0; k < P_; ++k) {
      const float* vrow = qkv + (long)(k * N_ + n) * (3 * E_) + 2 * E_ + h * DH_;
      a += wgt[k] * vrow[d];
    }
    o[(long)(q * N_ + n) * E_ + h * DH_ + d] = a;
  }
}

// ---------------- workspace layout (floats) ----------------
constexpr long o_sig    = 0;
constexpr long o_xfeat  = o_sig    + (long)N_ * P_ * HW_;
constexpr long o_pf     = o_xfeat  + (long)NP_ * C_;
constexpr long o_params = o_pf     + (long)R_ * C_;
constexpr long o_inp    = o_params + (long)NP_ * 2 * C_;
constexpr long o_gate   = o_inp    + (long)R_ * 2 * C_;
constexpr long o_tmp    = o_gate   + (long)R_ * C_;
constexpr long o_ig     = o_tmp    + (long)R_ * C_;
constexpr long o_ug     = o_ig     + (long)R_ * C_;
constexpr long o_pout   = o_ug     + (long)R_ * C_;
constexpr long o_iout   = o_pout   + (long)NP_ * C_;
constexpr long o_featc  = o_iout   + (long)R_ * C_;
constexpr long o_feat2  = o_featc  + (long)R_ * C_;
constexpr long o_obj    = o_feat2  + (long)R_ * C_;
constexpr long o_qkv    = o_obj    + (long)NP_ * E_;
constexpr long o_attno  = o_qkv    + (long)NP_ * 3 * E_;
constexpr long o_proj   = o_attno  + (long)NP_ * E_;
constexpr long o_obj2   = o_proj   + (long)NP_ * E_;
constexpr long o_objr   = o_obj2   + (long)NP_ * E_;
constexpr long o_h1     = o_objr   + (long)R_ * C_;
constexpr long o_h2     = o_h1     + (long)R_ * FFN_;
constexpr long o_obj3   = o_h2     + (long)R_ * C_;
constexpr long o_mfa    = o_obj3   + (long)R_ * C_;
constexpr long o_mfb    = o_mfa    + (long)R_ * C_;
constexpr long o_mfo    = o_mfb    + (long)R_ * C_;
constexpr long o_ker    = o_mfo    + (long)R_ * C_;

extern "C" void kernel_launch(void* const* d_in, const int* in_sizes, int n_in,
                              void* d_out, int out_size, void* d_ws, size_t ws_size,
                              hipStream_t stream) {
  (void)in_sizes; (void)n_in; (void)out_size; (void)ws_size;
  const float* x          = (const float*)d_in[0];
  const float* prop       = (const float*)d_in[1];
  const float* mp         = (const float*)d_in[2];
  const float* ku_dyn_w   = (const float*)d_in[3];
  const float* ku_dyn_b   = (const float*)d_in[4];
  const float* ku_in_w    = (const float*)d_in[5];
  const float* ku_in_b    = (const float*)d_in[6];
  const float* ku_ig_w    = (const float*)d_in[7];
  const float* ku_ig_b    = (const float*)d_in[8];
  const float* ku_ug_w    = (const float*)d_in[9];
  const float* ku_ug_b    = (const float*)d_in[10];
  const float* ku_nin_g   = (const float*)d_in[11];
  const float* ku_nin_b   = (const float*)d_in[12];
  const float* ku_nout_g  = (const float*)d_in[13];
  const float* ku_nout_b  = (const float*)d_in[14];
  const float* ku_inin_g  = (const float*)d_in[15];
  const float* ku_inin_b  = (const float*)d_in[16];
  const float* ku_inout_g = (const float*)d_in[17];
  const float* ku_inout_b = (const float*)d_in[18];
  const float* ku_fc_w    = (const float*)d_in[19];
  const float* ku_fc_b    = (const float*)d_in[20];
  const float* ku_fcn_g   = (const float*)d_in[21];
  const float* ku_fcn_b   = (const float*)d_in[22];
  const float* attn_in_w  = (const float*)d_in[23];
  const float* attn_in_b  = (const float*)d_in[24];
  const float* attn_out_w = (const float*)d_in[25];
  const float* attn_out_b = (const float*)d_in[26];
  const float* attn_ln_g  = (const float*)d_in[27];
  const float* attn_ln_b  = (const float*)d_in[28];
  const float* ffn_w1     = (const float*)d_in[29];
  const float* ffn_b1     = (const float*)d_in[30];
  const float* ffn_w2     = (const float*)d_in[31];
  const float* ffn_b2     = (const float*)d_in[32];
  const float* ffn_ln_g   = (const float*)d_in[33];
  const float* ffn_ln_b   = (const float*)d_in[34];
  const float* mask_fc_ws = (const float*)d_in[35];
  const float* mask_fc_ln_g = (const float*)d_in[36];
  const float* mask_fc_ln_b = (const float*)d_in[37];
  const float* fc_mask_w  = (const float*)d_in[38];
  const float* fc_mask_b  = (const float*)d_in[39];

  float* W = (float*)d_ws;
  float* out = (float*)d_out;
  const dim3 b256(256);
  const int gEW = (int)(((long)R_ * C_ + 255) / 256);  // elementwise grid (1800)

  auto gemm = [&](const float* A, long lda, long sA,
                  const float* B, long ldb, long sB, const float* bias,
                  float* Cp, long ldc, long sC,
                  int M, int Nc, int K, int act, int batch) {
    dim3 g(Nc / 32, (M + 63) / 64, batch);
    gemm_nt_wmma<<<g, b256, 0, stream>>>(A, lda, sA, B, ldb, sB, bias,
                                         Cp, ldc, sC, M, Nc, K, act);
  };

  // 1) softmax over P
  softmax_p_kernel<<<dim3(((long)N_ * HW_ + 255) / 256), b256, 0, stream>>>(mp, W + o_sig);

  // 2) x_feat[n] = sig[n] @ x[n]^T  (batched NT GEMM, K = HW)
  gemm(W + o_sig, HW_, (long)P_ * HW_, x, HW_, (long)C_ * HW_, nullptr,
       W + o_xfeat, C_, (long)P_ * C_, P_, C_, HW_, 0, N_);

  // 3) pf = proposal_feat (np, c, kk) -> (np, kk, c)
  ckk_to_kkc_kernel<<<gEW, b256, 0, stream>>>(prop, W + o_pf);

  // 4) params = upd @ ku_dyn_w^T + b   (200 x 512)
  gemm(W + o_xfeat, C_, 0, ku_dyn_w, C_, 0, ku_dyn_b,
       W + o_params, 2 * C_, 0, NP_, 2 * C_, C_, 0, 1);

  // 5) inp = pf @ ku_in_w^T + b        (1800 x 512)
  gemm(W + o_pf, C_, 0, ku_in_w, C_, 0, ku_in_b,
       W + o_inp, 2 * C_, 0, R_, 2 * C_, C_, 0, 1);

  // 6) gate = i_in * p_in
  gate_kernel<<<gEW, b256, 0, stream>>>(W + o_inp, W + o_params, W + o_gate);

  // 7) ig = sigmoid(LN(gate @ ku_ig^T + b ; inin))
  gemm(W + o_gate, C_, 0, ku_ig_w, C_, 0, ku_ig_b, W + o_tmp, C_, 0, R_, C_, C_, 0, 1);
  layernorm_kernel<<<R_, b256, 0, stream>>>(W + o_tmp, C_, nullptr, 0,
                                            ku_inin_g, ku_inin_b, W + o_ig, C_, C_, 1);
  // 8) ug = sigmoid(LN(gate @ ku_ug^T + b ; nin))
  gemm(W + o_gate, C_, 0, ku_ug_w, C_, 0, ku_ug_b, W + o_tmp, C_, 0, R_, C_, C_, 0, 1);
  layernorm_kernel<<<R_, b256, 0, stream>>>(W + o_tmp, C_, nullptr, 0,
                                            ku_nin_g, ku_nin_b, W + o_ug, C_, C_, 1);
  // 9) p_out = LN(params[:, C:]; nout) ; i_out = LN(inp[..., C:]; inout)
  layernorm_kernel<<<NP_, b256, 0, stream>>>(W + o_params + C_, 2 * C_, nullptr, 0,
                                             ku_nout_g, ku_nout_b, W + o_pout, C_, C_, 0);
  layernorm_kernel<<<R_, b256, 0, stream>>>(W + o_inp + C_, 2 * C_, nullptr, 0,
                                            ku_inout_g, ku_inout_b, W + o_iout, C_, C_, 0);
  // 10) feat = ug*p_out + ig*i_out ; feat = relu(LN(feat @ ku_fc^T + b ; fcn))
  feat_combine_kernel<<<gEW, b256, 0, stream>>>(W + o_ug, W + o_ig, W + o_pout,
                                                W + o_iout, W + o_featc);
  gemm(W + o_featc, C_, 0, ku_fc_w, C_, 0, ku_fc_b, W + o_tmp, C_, 0, R_, C_, C_, 0, 1);
  layernorm_kernel<<<R_, b256, 0, stream>>>(W + o_tmp, C_, nullptr, 0,
                                            ku_fcn_g, ku_fcn_b, W + o_feat2, C_, C_, 2);

  // 11) obj = feat.reshape(N,P,E).transpose(1,0,2)  -> (P,N,E) rows
  pack_obj_kernel<<<gEW, b256, 0, stream>>>(W + o_feat2, W + o_obj);

  // 12) qkv = obj @ attn_in_w^T + b   (200 x 6912, K = 2304)
  gemm(W + o_obj, E_, 0, attn_in_w, E_, 0, attn_in_b,
       W + o_qkv, 3 * E_, 0, NP_, 3 * E_, E_, 0, 1);

  // 13) multi-head attention
  attn_kernel<<<dim3(P_, NH_, N_), dim3(128), 0, stream>>>(W + o_qkv, W + o_attno);

  // 14) obj = LN(obj + o @ attn_out^T + b ; attn_ln)
  gemm(W + o_attno, E_, 0, attn_out_w, E_, 0, attn_out_b,
       W + o_proj, E_, 0, NP_, E_, E_, 0, 1);
  layernorm_kernel<<<NP_, b256, 0, stream>>>(W + o_proj, E_, W + o_obj, E_,
                                             attn_ln_g, attn_ln_b, W + o_obj2, E_, E_, 0);

  // 15) back to (N,P,KK,C) rows
  unpack_obj_kernel<<<gEW, b256, 0, stream>>>(W + o_obj2, W + o_objr);

  // 16) FFN: h = relu(objr @ w1^T + b1) @ w2^T + b2 ; obj3 = LN(objr + h)
  gemm(W + o_objr, C_, 0, ffn_w1, C_, 0, ffn_b1, W + o_h1, FFN_, 0, R_, FFN_, C_, 1, 1);
  gemm(W + o_h1, FFN_, 0, ffn_w2, FFN_, 0, ffn_b2, W + o_h2, C_, 0, R_, C_, FFN_, 0, 1);
  layernorm_kernel<<<R_, b256, 0, stream>>>(W + o_h2, C_, W + o_objr, C_,
                                            ffn_ln_g, ffn_ln_b, W + o_obj3, C_, C_, 0);

  // 17) mask-fc chain: mf = relu(LN(mf @ Wi^T)) x3, then fc_mask
  const float* mf_src = W + o_obj3;
  float* mf_dst[3] = {W + o_mfa, W + o_mfb, W + o_mfa};
  for (int i = 0; i < 3; ++i) {
    gemm(mf_src, C_, 0, mask_fc_ws + (long)i * C_ * C_, C_, 0, nullptr,
         W + o_tmp, C_, 0, R_, C_, C_, 0, 1);
    layernorm_kernel<<<R_, b256, 0, stream>>>(W + o_tmp, C_, nullptr, 0,
                                              mask_fc_ln_g + (long)i * C_,
                                              mask_fc_ln_b + (long)i * C_,
                                              mf_dst[i], C_, C_, 2);
    mf_src = mf_dst[i];
  }
  gemm(mf_src, C_, 0, fc_mask_w, C_, 0, fc_mask_b, W + o_mfo, C_, 0, R_, C_, C_, 0, 1);

  // 18) kernels = mf transposed to (N,P,C,KK)
  kkc_to_ckk_kernel<<<gEW, b256, 0, stream>>>(W + o_mfo, W + o_ker);

  // 19) new_masks = conv3x3(x, kernels)  -> d_out[0 : N*P*H*W]
  conv3x3_wmma<<<dim3(H_ * 3, (P_ + 31) / 32, N_), b256, 0, stream>>>(x, W + o_ker, out);

  // 20) obj_kernels = obj3 transposed to (N,P,C,K,K) -> d_out tail
  kkc_to_ckk_kernel<<<gEW, b256, 0, stream>>>(W + o_obj3, out + NM_);
}
